// AGNN_39041252720986
// MI455X (gfx1250) — matrix operand order repaired and checked by reading, not compiled, not run on hardware
//
#include <hip/hip_runtime.h>
#include <hip/hip_bf16.h>
#include <math.h>

// ---------------------------------------------------------------------------
// AGNN forward: h=relu(X@W1+b1); 2x AGNNConv(edge attention); out =
// log_softmax(h@W2+b2).  Dense GEMMs use v_wmma_f32_16x16x32_bf16 with
// bf16 operands staged in LDS in K-contiguous order so each fragment is
// exactly two ds_load_b128.  Edge phase: 128B row gathers + L2 atomics.
// ---------------------------------------------------------------------------

#define IN_SIZE  256
#define HID_SIZE 32
#define OUT_SIZE 64

typedef __attribute__((ext_vector_type(16))) __bf16         bf16x16;
typedef __attribute__((ext_vector_type(8)))  float          f32x8;
typedef __attribute__((ext_vector_type(16))) unsigned short u16x16;

union FragU { u16x16 u; bf16x16 b; uint4 q[2]; };

__device__ __forceinline__ unsigned short f32_to_bf16_rne(float f) {
    unsigned u = __float_as_uint(f);
    u += 0x7FFFu + ((u >> 16) & 1u);      // round to nearest even
    return (unsigned short)(u >> 16);
}

__device__ __forceinline__ uint2 pack4_bf16(float4 v) {
    unsigned lo = (unsigned)f32_to_bf16_rne(v.x) | ((unsigned)f32_to_bf16_rne(v.y) << 16);
    unsigned hi = (unsigned)f32_to_bf16_rne(v.z) | ((unsigned)f32_to_bf16_rne(v.w) << 16);
    return make_uint2(lo, hi);
}

// order-preserving float <-> uint map for atomic segment-max
__device__ __forceinline__ unsigned ordered_enc(float f) {
    unsigned u = __float_as_uint(f);
    return (u & 0x80000000u) ? ~u : (u | 0x80000000u);
}
__device__ __forceinline__ float ordered_dec(unsigned u) {
    unsigned v = (u & 0x80000000u) ? (u & 0x7FFFFFFFu) : ~u;
    return __uint_as_float(v);
}

// A fragment (16x32 bf16, row-major [m][k] in LDS): lane holds row m=lane&15;
// element e is K = hi*16 + e (16 contiguous bf16).
__device__ __forceinline__ FragU load_a_frag(const unsigned short* Arow /*[32] bf16*/,
                                             int hi) {
    FragU f;
    f.q[0] = *(const uint4*)(Arow + hi * 16);       // K run 0
    f.q[1] = *(const uint4*)(Arow + hi * 16 + 8);   // K run 1
    return f;
}

// B fragment (32x16 bf16, stored K-contiguous per column: Bt[n][k]): lane holds
// column n; element e is K = hi*16 + e  -> 16 contiguous bf16 = 2x b128.
__device__ __forceinline__ FragU load_b_frag(const unsigned short* Bcol /*[>=32] bf16*/,
                                             int hi) {
    FragU f;
    f.q[0] = *(const uint4*)(Bcol + hi * 16);
    f.q[1] = *(const uint4*)(Bcol + hi * 16 + 8);
    return f;
}

// ---------------------------------------------------------------------------
// Kernel 1: h = relu(features @ W1 + b1)   [N,256]x[256,32] -> [N,32]
// block = 256 threads = 8 waves; each wave owns a 16-row tile (block: 128 rows)
// ---------------------------------------------------------------------------
__global__ void gemm1_relu_kernel(const float* __restrict__ feats,
                                  const float* __restrict__ W1,
                                  const float* __restrict__ b1,
                                  float* __restrict__ h, int nNodes) {
    // W1 transposed to [n][k] bf16 (K-contiguous per output column): 16KB
    __shared__ __align__(16) unsigned short W1t[HID_SIZE * IN_SIZE];
    // per-wave 16x32 bf16 A tile, row-major [m][k]: 1KB x 8
    __shared__ __align__(16) unsigned short As[8][16 * 32];

    const int tid  = threadIdx.x;
    const int wave = tid >> 5;
    const int lane = tid & 31;
    const int m    = lane & 15;
    const int hi   = lane >> 4;
    const int row0 = blockIdx.x * 128 + wave * 16;

    // stage W1^T as bf16 (one-time; k contiguous per thread run)
    for (int i = tid; i < IN_SIZE * HID_SIZE; i += 256) {
        int k = i & (IN_SIZE - 1), n = i >> 8;
        W1t[n * IN_SIZE + k] = f32_to_bf16_rne(W1[k * HID_SIZE + n]);
    }

    f32x8 c0 = {}; f32x8 c1 = {};

    for (int k0 = 0; k0 < IN_SIZE; k0 += 32) {
        __syncthreads();   // covers W1t staging on first iteration, As reuse after
        // stage 16x32 A tile: load float4 coalesced, convert, store packed bf16
        #pragma unroll
        for (int i = 0; i < 4; ++i) {
            int idx = i * 32 + lane;           // 0..127
            int r = idx >> 3, c4 = (idx & 7) * 4;
            int gr = row0 + r;
            float4 v = make_float4(0.f, 0.f, 0.f, 0.f);
            if (gr < nNodes)
                v = *(const float4*)(feats + (size_t)gr * IN_SIZE + k0 + c4);
            *(uint2*)(&As[wave][r * 32 + c4]) = pack4_bf16(v);
        }
        __syncthreads();

        FragU af  = load_a_frag(&As[wave][m * 32], hi);
        FragU bf0 = load_b_frag(&W1t[m        * IN_SIZE + k0], hi);
        FragU bf1 = load_b_frag(&W1t[(16 + m) * IN_SIZE + k0], hi);

        c0 = __builtin_amdgcn_wmma_f32_16x16x32_bf16(false, af.b, false, bf0.b,
                                                     (short)0, c0, false, false);
        c1 = __builtin_amdgcn_wmma_f32_16x16x32_bf16(false, af.b, false, bf1.b,
                                                     (short)0, c1, false, false);
    }

    // D layout: lane holds column n=m; VGPR r -> row hi*8+r
    float bia0 = b1[m], bia1 = b1[16 + m];
    #pragma unroll
    for (int r = 0; r < 8; ++r) {
        int gm = row0 + hi * 8 + r;
        if (gm < nNodes) {
            float v0 = c0[r] + bia0;
            float v1 = c1[r] + bia1;
            h[(size_t)gm * HID_SIZE + m]      = v0 > 0.f ? v0 : 0.f;
            h[(size_t)gm * HID_SIZE + 16 + m] = v1 > 0.f ? v1 : 0.f;
        }
    }
}

// ---------------------------------------------------------------------------
// Row normalize xn = x / max(||x||,eps); also reset per-node max/sum state.
// 8 lanes per node (float4 each -> one 128B coalesced row load).
// ---------------------------------------------------------------------------
__global__ void normalize_kernel(const float* __restrict__ x, float* __restrict__ xn,
                                 unsigned* __restrict__ mmax, float* __restrict__ ssum,
                                 int nNodes) {
    int tid  = blockIdx.x * 256 + threadIdx.x;
    int node = tid >> 3, l8 = tid & 7;
    if (node >= nNodes) return;
    float4 v = *(const float4*)(x + (size_t)node * HID_SIZE + l8 * 4);
    float ss = v.x * v.x + v.y * v.y + v.z * v.z + v.w * v.w;
    ss += __shfl_xor(ss, 1, 32);
    ss += __shfl_xor(ss, 2, 32);
    ss += __shfl_xor(ss, 4, 32);
    float inv = 1.f / fmaxf(sqrtf(ss), 1e-12f);
    float4 o = make_float4(v.x * inv, v.y * inv, v.z * inv, v.w * inv);
    *(float4*)(xn + (size_t)node * HID_SIZE + l8 * 4) = o;
    if (l8 == 0) { mmax[node] = 0u; ssum[node] = 0.f; }  // 0 is below enc(x) for any non-NaN x
}

// ---------------------------------------------------------------------------
// Edge logits: beta * <xn[dst], xn[src]>; atomic segment-max over dst.
// ---------------------------------------------------------------------------
__global__ void logits_kernel(const int* __restrict__ src, const int* __restrict__ dst,
                              const float* __restrict__ xn, float* __restrict__ elog,
                              unsigned* __restrict__ mmax,
                              const float* __restrict__ betas, int layer, int nEdges) {
    int tid = blockIdx.x * 256 + threadIdx.x;
    int e = tid >> 3, l8 = tid & 7;
    if (e >= nEdges) return;
    int s = src[e], d = dst[e];
    float4 a = *(const float4*)(xn + (size_t)s * HID_SIZE + l8 * 4);
    float4 b = *(const float4*)(xn + (size_t)d * HID_SIZE + l8 * 4);
    float p = a.x * b.x + a.y * b.y + a.z * b.z + a.w * b.w;
    p += __shfl_xor(p, 1, 32);
    p += __shfl_xor(p, 2, 32);
    p += __shfl_xor(p, 4, 32);
    if (l8 == 0) {
        float lg = betas[layer] * p;
        elog[e] = lg;
        atomicMax(mmax + d, ordered_enc(lg));
    }
}

// ---------------------------------------------------------------------------
// e = exp(logit - m[dst]) (in place); atomic segment-sum over dst.
// ---------------------------------------------------------------------------
__global__ void expsum_kernel(const int* __restrict__ dst, float* __restrict__ elog,
                              const unsigned* __restrict__ mmax, float* __restrict__ ssum,
                              int nEdges) {
    int e = blockIdx.x * 256 + threadIdx.x;
    if (e >= nEdges) return;
    int d = dst[e];
    float v = expf(elog[e] - ordered_dec(mmax[d]));
    elog[e] = v;
    atomicAdd(ssum + d, v);
}

// ---------------------------------------------------------------------------
// agg[dst] += (e/s[dst]) * x[src]   (8 lanes per edge, 4 f32 atomics/lane)
// ---------------------------------------------------------------------------
__global__ void aggregate_kernel(const int* __restrict__ src, const int* __restrict__ dst,
                                 const float* __restrict__ elog, const float* __restrict__ ssum,
                                 const float* __restrict__ x, float* __restrict__ agg,
                                 int nEdges) {
    int tid = blockIdx.x * 256 + threadIdx.x;
    int e = tid >> 3, l8 = tid & 7;
    if (e >= nEdges) return;
    int s = src[e], d = dst[e];
    float alpha = elog[e] / ssum[d];
    float4 xv = *(const float4*)(x + (size_t)s * HID_SIZE + l8 * 4);
    float* o = agg + (size_t)d * HID_SIZE + l8 * 4;
    atomicAdd(o + 0, alpha * xv.x);
    atomicAdd(o + 1, alpha * xv.y);
    atomicAdd(o + 2, alpha * xv.z);
    atomicAdd(o + 3, alpha * xv.w);
}

__global__ void relu_copy_zero_kernel(float* __restrict__ hdst, float* __restrict__ agg, int n) {
    int i = blockIdx.x * 256 + threadIdx.x;
    if (i < n) { float v = agg[i]; hdst[i] = v > 0.f ? v : 0.f; agg[i] = 0.f; }
}

__global__ void zero_kernel(float* __restrict__ p, int n) {
    int i = blockIdx.x * 256 + threadIdx.x;
    if (i < n) p[i] = 0.f;
}

// ---------------------------------------------------------------------------
// out = log_softmax(h @ W2 + b2); wave computes 16x64 via 4 WMMAs, stages to
// LDS, then per-row LSE with a 16-lane pair reduction.
// ---------------------------------------------------------------------------
__global__ void gemm2_lsm_kernel(const float* __restrict__ h,
                                 const float* __restrict__ W2,
                                 const float* __restrict__ b2,
                                 float* __restrict__ out, int nNodes) {
    // W2 transposed to [n][k] bf16: 4KB
    __shared__ __align__(16) unsigned short W2t[OUT_SIZE * HID_SIZE];
    __shared__ __align__(16) unsigned short Hs[8][16 * 32];   // bf16 tiles 8KB
    __shared__ float Os[8][16 * 64];                          // 32KB

    const int tid  = threadIdx.x;
    const int wave = tid >> 5;
    const int lane = tid & 31;
    const int m    = lane & 15;
    const int hi   = lane >> 4;
    const int row0 = blockIdx.x * 128 + wave * 16;

    for (int i = tid; i < HID_SIZE * OUT_SIZE; i += 256) {
        int k = i & (HID_SIZE - 1), n = i >> 5;
        W2t[n * HID_SIZE + k] = f32_to_bf16_rne(W2[k * OUT_SIZE + n]);
    }

    #pragma unroll
    for (int i = 0; i < 4; ++i) {
        int idx = i * 32 + lane;
        int r = idx >> 3, c4 = (idx & 7) * 4;
        int gr = row0 + r;
        float4 v = make_float4(0.f, 0.f, 0.f, 0.f);
        if (gr < nNodes) v = *(const float4*)(h + (size_t)gr * HID_SIZE + c4);
        *(uint2*)(&Hs[wave][r * 32 + c4]) = pack4_bf16(v);
    }
    __syncthreads();

    FragU af = load_a_frag(&Hs[wave][m * 32], hi);

    #pragma unroll
    for (int nt = 0; nt < 4; ++nt) {
        FragU bf = load_b_frag(&W2t[(nt * 16 + m) * HID_SIZE], hi);
        f32x8 c = {};
        c = __builtin_amdgcn_wmma_f32_16x16x32_bf16(false, af.b, false, bf.b,
                                                    (short)0, c, false, false);
        float bia = b2[nt * 16 + m];
        #pragma unroll
        for (int r = 0; r < 8; ++r)
            Os[wave][(hi * 8 + r) * 64 + nt * 16 + m] = c[r] + bia;
    }
    __syncthreads();

    // lanes (m, m+16) handle row m: halves [0,32) and [32,64)
    float mx = -3.4e38f;
    for (int c = 0; c < 32; ++c) mx = fmaxf(mx, Os[wave][m * 64 + hi * 32 + c]);
    mx = fmaxf(mx, __shfl_xor(mx, 16, 32));
    float s = 0.f;
    for (int c = 0; c < 32; ++c) s += expf(Os[wave][m * 64 + hi * 32 + c] - mx);
    s += __shfl_xor(s, 16, 32);
    float lse = mx + logf(s);
    int gr = row0 + m;
    if (gr < nNodes) {
        for (int c = 0; c < 32; ++c)
            out[(size_t)gr * OUT_SIZE + hi * 32 + c] = Os[wave][m * 64 + hi * 32 + c] - lse;
    }
}

// ---------------------------------------------------------------------------
extern "C" void kernel_launch(void* const* d_in, const int* in_sizes, int n_in,
                              void* d_out, int out_size, void* d_ws, size_t ws_size,
                              hipStream_t stream) {
    const int*   edge  = (const int*)d_in[0];     // [2, E] int32
    const float* feats = (const float*)d_in[1];   // [N, 256]
    const float* W1    = (const float*)d_in[2];   // [256, 32]
    const float* b1    = (const float*)d_in[3];   // [32]
    const float* betas = (const float*)d_in[4];   // [2]
    const float* W2    = (const float*)d_in[5];   // [32, 64]
    const float* b2    = (const float*)d_in[6];   // [64]
    float*       out   = (float*)d_out;

    const int E = in_sizes[0] / 2;
    const int N = in_sizes[1] / IN_SIZE;
    const int* src = edge;
    const int* dst = edge + E;

    // workspace layout (all 16B aligned)
    char* ws = (char*)d_ws;
    const size_t NB = (size_t)N * HID_SIZE * sizeof(float);
    float*    h0   = (float*)(ws);
    float*    xn   = (float*)(ws + NB);
    float*    agg  = (float*)(ws + 2 * NB);
    unsigned* mmax = (unsigned*)(ws + 3 * NB);
    float*    ssum = (float*)(ws + 3 * NB + (size_t)N * 4);
    float*    elog = (float*)(ws + 3 * NB + (size_t)N * 8);

    const int gRows  = (N + 127) / 128;          // 128 rows per block (8 waves x 16)
    const int gNodeF = ((size_t)N * HID_SIZE + 255) / 256;
    const int gNode8 = ((size_t)N * 8 + 255) / 256;
    const int gEdge8 = ((size_t)E * 8 + 255) / 256;
    const int gEdge  = (E + 255) / 256;

    gemm1_relu_kernel<<<gRows, 256, 0, stream>>>(feats, W1, b1, h0, N);
    zero_kernel<<<gNodeF, 256, 0, stream>>>(agg, N * HID_SIZE);

    for (int layer = 0; layer < 2; ++layer) {
        normalize_kernel<<<gNode8, 256, 0, stream>>>(h0, xn, mmax, ssum, N);
        logits_kernel<<<gEdge8, 256, 0, stream>>>(src, dst, xn, elog, mmax, betas, layer, E);
        expsum_kernel<<<gEdge, 256, 0, stream>>>(dst, elog, mmax, ssum, E);
        aggregate_kernel<<<gEdge8, 256, 0, stream>>>(src, dst, elog, ssum, h0, agg, E);
        relu_copy_zero_kernel<<<gNodeF, 256, 0, stream>>>(h0, agg, N * HID_SIZE);
    }

    gemm2_lsm_kernel<<<gRows, 256, 0, stream>>>(h0, W2, b2, out, N);
}
